// Model_69810398429871
// MI455X (gfx1250) — compile-verified, matrix operand order checked by reading, not run on hardware
//
#include <hip/hip_runtime.h>
#include <hip/hip_bf16.h>
#include <stdint.h>

typedef __attribute__((ext_vector_type(16))) _Float16 v16h;
typedef __attribute__((ext_vector_type(8)))  _Float16 v8h;
typedef __attribute__((ext_vector_type(4)))  _Float16 v4h;
typedef __attribute__((ext_vector_type(8)))  float    v8f;

#define NATOMS 22
#define NEDGE  484   // 22*22
#define NEPAD  496   // 31 tiles * 16
#define HDIM   128
#define MM     3
#define LL     4
#define BB     512

// ---- workspace byte offsets (f16 packed transposed weights, etc.) ----
#define OFF_EMBT   0u                       // M * [128n][32k] f16
#define OFF_E1AT   24576u                   // M*L * [128][128]
#define OFF_E1BT   417792u
#define OFF_E2T    811008u
#define OFF_N1AT   1204224u
#define OFF_N1BT   1597440u
#define OFF_N1CT   1990656u                 // M*L * [128n][32k]
#define OFF_N2T    2088960u
#define OFF_D1T    2482176u                 // M * [128][128]
#define OFF_D2T    2580480u
#define OFF_W1R    2678784u                 // M*L*128 f32 (radial row of ew1)
#define OFF_PREDS  2684928u                 // M*B f32
#define WS_NEEDED  2691072u

// ---- LDS pool byte offsets (single block, manual overlay) ----
#define LDS_HF     0u        //  23x128 f16  h (row 22 = dump row)   5888
#define LDS_H0F    5888u     //  32x32  f16  h0 padded               2048
#define LDS_RE     7936u     //  496 f32 radial per edge             1984
#define LDS_MASK   9920u     //  496 f32 edge mask (0 beyond 484)    1984
#define LDS_W1R    11904u    //  128 f32                              512
#define LDS_RED    12416u    //  128 f32 pooled                       512
#define LDS_V      12928u    //  128 f32 graph-dec tmp                512
#define LDS_X      13440u    //  66 f32 coords (pad 272)              272
#define LDS_IJ     13712u    //  496 u16 packed (i<<8|j)              992
#define LDS_AGG    14704u    //  23x128 f32 (row 22 = dump)         11776
#define LDS_P      26480u    //  23x128 f32                         11776
#define LDS_Q      38256u    //  23x128 f32                         11776
#define LDS_T0     54128u    //  16x128 f16                          4096
#define LDS_T1     58224u    //  16x128 f16                          4096
#define LDS_AGGF   26480u    //  23x128 f16 (overlays sP)            5888
#define LDS_U      32368u    //  23x128 f16 (overlays sP)            5888
#define LDS_BYTES  62320u

__device__ __forceinline__ float silu_f(float x) {
    return x / (1.0f + __expf(-x));
}

// 16-row x 32-K 16-bit operand fragment (A or B role) from row-major [rows][ld]
// f16 array: lane&15 -> row, lane&16 -> +8 / +16..31 K half, VGPR v = K pair.
__device__ __forceinline__ v16h load_frag16(const _Float16* base, int row0,
                                            int k0, int ld, int lane) {
    const _Float16* p = base + (size_t)(row0 + (lane & 15)) * ld + k0
                        + ((lane & 16) ? 8 : 0);
    union { v16h v; v8h h[2]; } u;
    u.h[0] = *(const v8h*)(p);
    u.h[1] = *(const v8h*)(p + 16);
    return u.v;
}

#define WMMA_F16(A, B, C) \
    __builtin_amdgcn_wmma_f32_16x16x32_f16(false, (A), false, (B), (short)0, (C), false, false)

// ---------------------------------------------------------------------------
// Weight packing: f32 -> f16, transposed to [N][K] (K contiguous) per matrix.
// ---------------------------------------------------------------------------
__global__ void k_pack(const float* __restrict__ emb_w, const float* __restrict__ ew1,
                       const float* __restrict__ ew2,  const float* __restrict__ nw1,
                       const float* __restrict__ nw2,  const float* __restrict__ dw1,
                       const float* __restrict__ dw2,
                       _Float16* embT, _Float16* e1aT, _Float16* e1bT, _Float16* e2T,
                       _Float16* n1aT, _Float16* n1bT, _Float16* n1cT, _Float16* n2T,
                       _Float16* d1T,  _Float16* d2T,  float* w1r) {
    int t0 = blockIdx.x * blockDim.x + threadIdx.x;
    int st = gridDim.x * blockDim.x;
    for (int idx = t0; idx < MM * 128 * 32; idx += st) {
        int m = idx / (128 * 32), rem = idx % (128 * 32);
        int n = rem / 32, k = rem % 32;
        embT[idx] = (k < 15) ? (_Float16)emb_w[((size_t)m * 15 + k) * 128 + n]
                             : (_Float16)0.0f;
    }
    for (int idx = t0; idx < 12 * 16384; idx += st) {
        int mat = idx / 16384, rem = idx & 16383;
        int n = rem >> 7, k = rem & 127;
        e1aT[idx] = (_Float16)ew1[((size_t)mat * 257 + k) * 128 + n];
        e1bT[idx] = (_Float16)ew1[((size_t)mat * 257 + 128 + k) * 128 + n];
        e2T[idx]  = (_Float16)ew2[((size_t)mat * 128 + k) * 128 + n];
        n1aT[idx] = (_Float16)nw1[((size_t)mat * 271 + k) * 128 + n];
        n1bT[idx] = (_Float16)nw1[((size_t)mat * 271 + 128 + k) * 128 + n];
        n2T[idx]  = (_Float16)nw2[((size_t)mat * 128 + k) * 128 + n];
    }
    for (int idx = t0; idx < 12 * 128 * 32; idx += st) {
        int mat = idx / (128 * 32), rem = idx % (128 * 32);
        int n = rem / 32, k = rem % 32;
        n1cT[idx] = (k < 15)
            ? (_Float16)nw1[((size_t)mat * 271 + 256 + k) * 128 + n]
            : (_Float16)0.0f;
    }
    for (int idx = t0; idx < MM * 16384; idx += st) {
        int mat = idx / 16384, rem = idx & 16383;
        int n = rem >> 7, k = rem & 127;
        d1T[idx] = (_Float16)dw1[((size_t)mat * 128 + k) * 128 + n];
        d2T[idx] = (_Float16)dw2[((size_t)mat * 128 + k) * 128 + n];
    }
    for (int idx = t0; idx < 12 * 128; idx += st) {
        int mat = idx >> 7, n = idx & 127;
        w1r[idx] = ew1[((size_t)mat * 257 + 256) * 128 + n];
    }
}

// ---------------------------------------------------------------------------
// Fused EGNN: one workgroup per (molecule b, member m); 8 waves = 8 N-tiles.
// ---------------------------------------------------------------------------
__global__ __launch_bounds__(256) void k_egnn(
    const float* __restrict__ x, const float* __restrict__ h0,
    const float* __restrict__ node_mask, const float* __restrict__ edge_mask,
    const float* __restrict__ emb_b, const float* __restrict__ eb1,
    const float* __restrict__ eb2,  const float* __restrict__ nb1,
    const float* __restrict__ nb2,  const float* __restrict__ db1,
    const float* __restrict__ db2,  const float* __restrict__ gw1,
    const float* __restrict__ gb1,  const float* __restrict__ gw2,
    const float* __restrict__ gb2,
    const _Float16* __restrict__ embT, const _Float16* __restrict__ e1aT,
    const _Float16* __restrict__ e1bT, const _Float16* __restrict__ e2T,
    const _Float16* __restrict__ n1aT, const _Float16* __restrict__ n1bT,
    const _Float16* __restrict__ n1cT, const _Float16* __restrict__ n2T,
    const _Float16* __restrict__ d1T,  const _Float16* __restrict__ d2T,
    const float* __restrict__ w1rB, float* __restrict__ preds) {

    const int b    = blockIdx.x;
    const int m    = blockIdx.y;
    const int tid  = threadIdx.x;
    const int lane = tid & 31;
    const int n0   = (tid >> 5) * 16;
    const int ncol = n0 + (lane & 15);
    const int rofs = (lane & 16) ? 8 : 0;

    __shared__ __align__(16) unsigned char smem[LDS_BYTES];
    _Float16* sHf   = (_Float16*)(smem + LDS_HF);
    _Float16* sH0f  = (_Float16*)(smem + LDS_H0F);
    float*    sRe   = (float*)(smem + LDS_RE);
    float*    sMask = (float*)(smem + LDS_MASK);
    float*    sW1r  = (float*)(smem + LDS_W1R);
    float*    sRed  = (float*)(smem + LDS_RED);
    float*    sV    = (float*)(smem + LDS_V);
    float*    sX    = (float*)(smem + LDS_X);
    unsigned short* sIJ = (unsigned short*)(smem + LDS_IJ);
    float*    sAgg  = (float*)(smem + LDS_AGG);
    float*    sP    = (float*)(smem + LDS_P);
    float*    sQ    = (float*)(smem + LDS_Q);
    _Float16* sT0   = (_Float16*)(smem + LDS_T0);
    _Float16* sT1   = (_Float16*)(smem + LDS_T1);
    _Float16* sAggF = (_Float16*)(smem + LDS_AGGF);
    _Float16* sU    = (_Float16*)(smem + LDS_U);

    float hreg[16];               // f32 master h, resident in registers

    // ---- setup: coords, h0(f16), per-edge tables ----
    for (int i = tid; i < NATOMS * 3; i += 256) sX[i] = x[(size_t)b * NATOMS * 3 + i];
    for (int i = tid; i < NATOMS * 32; i += 256) {
        int r = i >> 5, k = i & 31;
        sH0f[i] = (k < 15) ? (_Float16)h0[((size_t)b * NATOMS + r) * 15 + k]
                           : (_Float16)0.0f;
    }
    __syncthreads();
    {
        const float* emask = edge_mask + (size_t)b * NEDGE;
        for (int e = tid; e < NEPAD; e += 256) {
            if (e < NEDGE) {
                int ii = e / NATOMS, jj = e - ii * NATOMS;
                float dx = sX[ii * 3 + 0] - sX[jj * 3 + 0];
                float dy = sX[ii * 3 + 1] - sX[jj * 3 + 1];
                float dz = sX[ii * 3 + 2] - sX[jj * 3 + 2];
                sRe[e]   = dx * dx + dy * dy + dz * dz;
                sMask[e] = emask[e];
                sIJ[e]   = (unsigned short)((ii << 8) | jj);
            } else {
                sRe[e] = 0.0f; sMask[e] = 0.0f;
                sIJ[e] = (unsigned short)(22 << 8);
            }
        }
    }
    // ---- embedding: h = h0 @ emb_w + emb_b ----
    {
        const _Float16* wemb = embT + (size_t)m * 128 * 32;
        v8f a0c = {}, a1c = {};
        v16h w  = load_frag16(wemb, n0, 0, 32, lane);
        v16h a0 = load_frag16(sH0f, 0, 0, 32, lane);
        v16h a1 = load_frag16(sH0f, 16, 0, 32, lane);
        a0c = WMMA_F16(a0, w, a0c);
        a1c = WMMA_F16(a1, w, a1c);
        float bias = emb_b[(size_t)m * HDIM + ncol];
#pragma unroll
        for (int r = 0; r < 8; ++r) {
            int r0 = r + rofs, r1 = 16 + r + rofs;
            float v0 = a0c[r] + bias, v1 = a1c[r] + bias;
            hreg[r] = v0; hreg[8 + r] = v1;
            sHf[((r0 < 22) ? r0 : 22) * HDIM + ncol] = (_Float16)v0;
            sHf[((r1 < 22) ? r1 : 22) * HDIM + ncol] = (_Float16)v1;
        }
    }
    __syncthreads();

    // ---- layers ----
    for (int l = 0; l < LL; ++l) {
        const size_t ml = (size_t)(m * LL + l);
        const _Float16* w1a = e1aT + ml * 16384;
        const _Float16* w1b = e1bT + ml * 16384;
        const _Float16* w2  = e2T  + ml * 16384;
        const _Float16* wna = n1aT + ml * 16384;
        const _Float16* wnb = n1bT + ml * 16384;
        const _Float16* wnc = n1cT + ml * 4096;
        const _Float16* wn2 = n2T  + ml * 16384;
        if (tid < HDIM) sW1r[tid] = w1rB[ml * HDIM + tid];
        __syncthreads();

        // E1: P = h@W1a + b1, Q = h@W1b (all-pairs factorization)
        {
            v8f p0 = {}, p1 = {}, q0 = {}, q1 = {};
#pragma unroll
            for (int kk = 0; kk < 4; ++kk) {
                v16h a0 = load_frag16(sHf, 0,  kk * 32, HDIM, lane);
                v16h a1 = load_frag16(sHf, 16, kk * 32, HDIM, lane);
                v16h bp = load_frag16(w1a, n0, kk * 32, HDIM, lane);
                v16h bq = load_frag16(w1b, n0, kk * 32, HDIM, lane);
                p0 = WMMA_F16(a0, bp, p0);  p1 = WMMA_F16(a1, bp, p1);
                q0 = WMMA_F16(a0, bq, q0);  q1 = WMMA_F16(a1, bq, q1);
            }
            float bias1 = eb1[ml * HDIM + ncol];
#pragma unroll
            for (int r = 0; r < 8; ++r) {
                int r0 = r + rofs, r1 = 16 + r + rofs;
                int c0 = ((r0 < 22) ? r0 : 22) * HDIM + ncol;
                int c1 = ((r1 < 22) ? r1 : 22) * HDIM + ncol;
                sP[c0] = p0[r] + bias1;  sQ[c0] = q0[r];
                sP[c1] = p1[r] + bias1;  sQ[c1] = q1[r];
            }
        }
        __syncthreads();
        for (int i = tid; i < (23 * HDIM) / 4; i += 256)
            *(float4*)&sAgg[i << 2] = make_float4(0.f, 0.f, 0.f, 0.f);

        // E2: msg = silu(silu(P_i+Q_j+r*w1r) @ W2 + b2)*mask ; agg_i += msg
        {
            // prefetch node-MLP weights (used after E2) into cache, overlapped
            __builtin_prefetch((const void*)(wna + (size_t)tid * 64), 0, 1);
            __builtin_prefetch((const void*)(wnb + (size_t)tid * 64), 0, 1);
            __builtin_prefetch((const void*)(wn2 + (size_t)tid * 64), 0, 1);

            v16h bw[4];
#pragma unroll
            for (int kk = 0; kk < 4; ++kk)
                bw[kk] = load_frag16(w2, n0, kk * 32, HDIM, lane);
            float bias2 = eb2[ml * HDIM + ncol];

            // build 16x128 f16 t-tile, 4 elements/thread/iter (float4 in,
            // v4h out) -> ds_load_b128 / ds_store_b64
            auto buildTile = [&](int et, _Float16* dst) {
                const int e0 = et * 16;
                for (int idx = tid; idx < 512; idx += 256) {
                    int r = idx >> 5, kq = (idx & 31) << 2;
                    int e = e0 + r;
                    int ij = sIJ[e];
                    const float4 p4 = *(const float4*)&sP[(ij >> 8) * HDIM + kq];
                    const float4 q4 = *(const float4*)&sQ[(ij & 255) * HDIM + kq];
                    const float4 w4 = *(const float4*)&sW1r[kq];
                    float re = sRe[e];
                    v4h o;
                    o[0] = (_Float16)silu_f(p4.x + q4.x + re * w4.x);
                    o[1] = (_Float16)silu_f(p4.y + q4.y + re * w4.y);
                    o[2] = (_Float16)silu_f(p4.z + q4.z + re * w4.z);
                    o[3] = (_Float16)silu_f(p4.w + q4.w + re * w4.w);
                    *(v4h*)(dst + (idx << 2)) = o;
                }
            };
            buildTile(0, sT0);
            __syncthreads();
            for (int et = 0; et < 31; ++et) {
                const _Float16* cur = (et & 1) ? sT1 : sT0;
                _Float16*       nxt = (et & 1) ? sT0 : sT1;
                v16h a0 = load_frag16(cur, 0, 0,  HDIM, lane);
                v16h a1 = load_frag16(cur, 0, 32, HDIM, lane);
                v16h a2 = load_frag16(cur, 0, 64, HDIM, lane);
                v16h a3 = load_frag16(cur, 0, 96, HDIM, lane);
                if (et + 1 < 31) buildTile(et + 1, nxt);   // overlap with WMMA
                v8f acc = {};
                acc = WMMA_F16(a0, bw[0], acc);
                acc = WMMA_F16(a1, bw[1], acc);
                acc = WMMA_F16(a2, bw[2], acc);
                acc = WMMA_F16(a3, bw[3], acc);
                // segment-sum: i = e/22 changes at most once over 8 edges ->
                // register pre-reduction into two runs, <=2 LDS atomics.
                const int eBase = et * 16 + rofs;
                const int ii0 = sIJ[eBase] >> 8;
                const int ii7 = sIJ[eBase + 7] >> 8;
                float s0 = 0.0f, s1 = 0.0f;
#pragma unroll
                for (int r = 0; r < 8; ++r) {
                    int e = eBase + r;
                    float msg = silu_f(acc[r] + bias2) * sMask[e];
                    bool lo = ((sIJ[e] >> 8) == ii0);
                    s0 += lo ? msg : 0.0f;
                    s1 += lo ? 0.0f : msg;
                }
                atomicAdd(&sAgg[ii0 * HDIM + ncol], s0);
                if (ii7 != ii0) atomicAdd(&sAgg[ii7 * HDIM + ncol], s1);
                __syncthreads();
            }
        }

        // prefetch next layer's edge weights, overlapped with node stage
        if (l + 1 < LL) {
            __builtin_prefetch((const void*)(w1a + 16384 + (size_t)tid * 64), 0, 1);
            __builtin_prefetch((const void*)(w1b + 16384 + (size_t)tid * 64), 0, 1);
            __builtin_prefetch((const void*)(w2  + 16384 + (size_t)tid * 64), 0, 1);
        }
        // node update: U = silu(h@Wna + agg@Wnb + h0@Wnc + nb1)
        for (int idx = tid; idx < (NATOMS * HDIM) / 4; idx += 256) {
            const float4 a4 = *(const float4*)&sAgg[idx << 2];
            v4h o;
            o[0] = (_Float16)a4.x; o[1] = (_Float16)a4.y;
            o[2] = (_Float16)a4.z; o[3] = (_Float16)a4.w;
            *(v4h*)(sAggF + (idx << 2)) = o;
        }
        __syncthreads();
        {
            v8f c0 = {}, c1 = {};
#pragma unroll
            for (int kk = 0; kk < 4; ++kk) {
                v16h w  = load_frag16(wna, n0, kk * 32, HDIM, lane);
                v16h a0 = load_frag16(sHf, 0,  kk * 32, HDIM, lane);
                v16h a1 = load_frag16(sHf, 16, kk * 32, HDIM, lane);
                c0 = WMMA_F16(a0, w, c0);  c1 = WMMA_F16(a1, w, c1);
            }
#pragma unroll
            for (int kk = 0; kk < 4; ++kk) {
                v16h w  = load_frag16(wnb, n0, kk * 32, HDIM, lane);
                v16h a0 = load_frag16(sAggF, 0,  kk * 32, HDIM, lane);
                v16h a1 = load_frag16(sAggF, 16, kk * 32, HDIM, lane);
                c0 = WMMA_F16(a0, w, c0);  c1 = WMMA_F16(a1, w, c1);
            }
            {
                v16h w  = load_frag16(wnc, n0, 0, 32, lane);
                v16h a0 = load_frag16(sH0f, 0,  0, 32, lane);
                v16h a1 = load_frag16(sH0f, 16, 0, 32, lane);
                c0 = WMMA_F16(a0, w, c0);  c1 = WMMA_F16(a1, w, c1);
            }
            float bias = nb1[ml * HDIM + ncol];
#pragma unroll
            for (int r = 0; r < 8; ++r) {
                int r0 = r + rofs, r1 = 16 + r + rofs;
                sU[((r0 < 22) ? r0 : 22) * HDIM + ncol] = (_Float16)silu_f(c0[r] + bias);
                sU[((r1 < 22) ? r1 : 22) * HDIM + ncol] = (_Float16)silu_f(c1[r] + bias);
            }
        }
        __syncthreads();
        // h += U @ Wn2 + nb2 (residual, h in registers)
        {
            v8f c0 = {}, c1 = {};
#pragma unroll
            for (int kk = 0; kk < 4; ++kk) {
                v16h w  = load_frag16(wn2, n0, kk * 32, HDIM, lane);
                v16h a0 = load_frag16(sU, 0,  kk * 32, HDIM, lane);
                v16h a1 = load_frag16(sU, 16, kk * 32, HDIM, lane);
                c0 = WMMA_F16(a0, w, c0);  c1 = WMMA_F16(a1, w, c1);
            }
            float bias = nb2[ml * HDIM + ncol];
#pragma unroll
            for (int r = 0; r < 8; ++r) {
                int r0 = r + rofs, r1 = 16 + r + rofs;
                float v0 = hreg[r]     + c0[r] + bias;
                float v1 = hreg[8 + r] + c1[r] + bias;
                hreg[r] = v0; hreg[8 + r] = v1;
                sHf[((r0 < 22) ? r0 : 22) * HDIM + ncol] = (_Float16)v0;
                sHf[((r1 < 22) ? r1 : 22) * HDIM + ncol] = (_Float16)v1;
            }
        }
        __syncthreads();
    }

    // ---- node decoder + pooling + graph decoder ----
    float colsum = 0.0f;
    {
        const _Float16* pd1 = d1T + (size_t)m * 16384;
        const _Float16* pd2 = d2T + (size_t)m * 16384;
        v8f c0 = {}, c1 = {};
#pragma unroll
        for (int kk = 0; kk < 4; ++kk) {
            v16h w  = load_frag16(pd1, n0, kk * 32, HDIM, lane);
            v16h a0 = load_frag16(sHf, 0,  kk * 32, HDIM, lane);
            v16h a1 = load_frag16(sHf, 16, kk * 32, HDIM, lane);
            c0 = WMMA_F16(a0, w, c0);  c1 = WMMA_F16(a1, w, c1);
        }
        float b1 = db1[(size_t)m * HDIM + ncol];
#pragma unroll
        for (int r = 0; r < 8; ++r) {
            int r0 = r + rofs, r1 = 16 + r + rofs;
            sU[((r0 < 22) ? r0 : 22) * HDIM + ncol] = (_Float16)silu_f(c0[r] + b1);
            sU[((r1 < 22) ? r1 : 22) * HDIM + ncol] = (_Float16)silu_f(c1[r] + b1);
        }
        __syncthreads();
        v8f d0 = {}, d1v = {};
#pragma unroll
        for (int kk = 0; kk < 4; ++kk) {
            v16h w  = load_frag16(pd2, n0, kk * 32, HDIM, lane);
            v16h a0 = load_frag16(sU, 0,  kk * 32, HDIM, lane);
            v16h a1 = load_frag16(sU, 16, kk * 32, HDIM, lane);
            d0 = WMMA_F16(a0, w, d0);  d1v = WMMA_F16(a1, w, d1v);
        }
        float b2 = db2[(size_t)m * HDIM + ncol];
        const float* nmk = node_mask + (size_t)b * NATOMS;
#pragma unroll
        for (int r = 0; r < 8; ++r) {
            int r0 = r + rofs, r1 = 16 + r + rofs;
            float nm0 = nmk[(r0 < 22) ? r0 : 0];
            float nm1 = nmk[(r1 < 22) ? r1 : 0];
            colsum += (r0 < 22) ? (d0[r]  + b2) * nm0 : 0.0f;
            colsum += (r1 < 22) ? (d1v[r] + b2) * nm1 : 0.0f;
        }
    }
    colsum += __shfl_xor(colsum, 16);      // combine row halves within wave
    if (lane < 16) sRed[ncol] = colsum;
    __syncthreads();
    if (tid < HDIM) {
        const float* g1 = gw1 + (size_t)m * HDIM * HDIM;
        float acc = gb1[(size_t)m * HDIM + tid];
        for (int k = 0; k < HDIM; ++k) acc += sRed[k] * g1[(size_t)k * HDIM + tid];
        sV[tid] = silu_f(acc) * gw2[(size_t)m * HDIM + tid];
    }
    __syncthreads();
    if (tid == 0) {
        float s = gb2[m];
        for (int n = 0; n < HDIM; ++n) s += sV[n];
        preds[(size_t)m * BB + b] = s;
    }
}

__global__ void k_mean(const float* __restrict__ preds, float* __restrict__ out) {
    int i = blockIdx.x * blockDim.x + threadIdx.x;
    if (i < BB)
        out[i] = (preds[i] + preds[BB + i] + preds[2 * BB + i]) * (1.0f / 3.0f);
}

extern "C" void kernel_launch(void* const* d_in, const int* in_sizes, int n_in,
                              void* d_out, int out_size, void* d_ws, size_t ws_size,
                              hipStream_t stream) {
    (void)in_sizes; (void)n_in; (void)out_size;
    if (ws_size < (size_t)WS_NEEDED) return;
    uint8_t* ws = (uint8_t*)d_ws;

    const float* x         = (const float*)d_in[0];
    const float* h0        = (const float*)d_in[1];
    const float* node_mask = (const float*)d_in[4];
    const float* edge_mask = (const float*)d_in[5];
    const float* emb_w = (const float*)d_in[7];
    const float* emb_b = (const float*)d_in[8];
    const float* ew1   = (const float*)d_in[9];
    const float* eb1   = (const float*)d_in[10];
    const float* ew2   = (const float*)d_in[11];
    const float* eb2   = (const float*)d_in[12];
    const float* nw1   = (const float*)d_in[13];
    const float* nb1   = (const float*)d_in[14];
    const float* nw2   = (const float*)d_in[15];
    const float* nb2   = (const float*)d_in[16];
    const float* dw1   = (const float*)d_in[17];
    const float* db1   = (const float*)d_in[18];
    const float* dw2   = (const float*)d_in[19];
    const float* db2   = (const float*)d_in[20];
    const float* gw1   = (const float*)d_in[21];
    const float* gb1   = (const float*)d_in[22];
    const float* gw2   = (const float*)d_in[23];
    const float* gb2   = (const float*)d_in[24];

    _Float16* embT = (_Float16*)(ws + OFF_EMBT);
    _Float16* e1aT = (_Float16*)(ws + OFF_E1AT);
    _Float16* e1bT = (_Float16*)(ws + OFF_E1BT);
    _Float16* e2T  = (_Float16*)(ws + OFF_E2T);
    _Float16* n1aT = (_Float16*)(ws + OFF_N1AT);
    _Float16* n1bT = (_Float16*)(ws + OFF_N1BT);
    _Float16* n1cT = (_Float16*)(ws + OFF_N1CT);
    _Float16* n2T  = (_Float16*)(ws + OFF_N2T);
    _Float16* d1T  = (_Float16*)(ws + OFF_D1T);
    _Float16* d2T  = (_Float16*)(ws + OFF_D2T);
    float*    w1rB = (float*)(ws + OFF_W1R);
    float*    prd  = (float*)(ws + OFF_PREDS);

    k_pack<<<256, 256, 0, stream>>>(emb_w, ew1, ew2, nw1, nw2, dw1, dw2,
                                    embT, e1aT, e1bT, e2T, n1aT, n1bT, n1cT,
                                    n2T, d1T, d2T, w1rB);

    dim3 grid(BB, MM);
    k_egnn<<<grid, 256, 0, stream>>>(x, h0, node_mask, edge_mask,
                                     emb_b, eb1, eb2, nb1, nb2, db1, db2,
                                     gw1, gb1, gw2, gb2,
                                     embT, e1aT, e1bT, e2T, n1aT, n1bT, n1cT,
                                     n2T, d1T, d2T, w1rB, prd);

    k_mean<<<2, 256, 0, stream>>>(prd, (float*)d_out);
}
